// LinearAttention_4148938408188
// MI455X (gfx1250) — matrix operand order repaired and checked by reading, not compiled, not run on hardware
//
#include <hip/hip_runtime.h>
#include <math.h>

#define B_    16
#define DIMC  256
#define NPOS  8192
#define HEADS 4
#define DHEAD 32
#define HID   128   // HEADS*DHEAD
#define O3    384   // 3*HID

typedef __attribute__((ext_vector_type(16))) __bf16 v16bf;
typedef __attribute__((ext_vector_type(8)))  float  v8f;

union BF16x16 {
    uint4          q[2];
    unsigned short u[16];
    v16bf          v;
};

__device__ __forceinline__ unsigned short f2bf(float f) {
    unsigned u = __float_as_uint(f);
    unsigned r = u + 0x7FFFu + ((u >> 16) & 1u);   // round-to-nearest-even
    return (unsigned short)(r >> 16);
}
__device__ __forceinline__ float bf2f(unsigned short s) {
    return __uint_as_float(((unsigned)s) << 16);
}

// CDNA5 async global->LDS copy (16B per lane), tracked by ASYNCcnt.
// LDS operand = low 32 bits of the flat shared pointer (flat LDS aperture
// keeps the byte offset in addr[31:0]).
__device__ __forceinline__ void async_copy16(const void* g, void* l) {
    unsigned lds = (unsigned)(size_t)l;
    asm volatile("global_load_async_to_lds_b128 %0, %1, off"
                 :: "v"(lds), "v"(g) : "memory");
}
__device__ __forceinline__ void wait_async0() {
    asm volatile("s_wait_asynccnt 0x0" ::: "memory");
}

// ---------------------------------------------------------------- kernel 0
__global__ __launch_bounds__(256) void cvt_weights(
    const float* __restrict__ wqkv, const float* __restrict__ wout,
    unsigned short* __restrict__ wqkv_bf, unsigned short* __restrict__ wout_bf) {
    int i = blockIdx.x * 256 + threadIdx.x;
    if (i < O3 * DIMC)  wqkv_bf[i] = f2bf(wqkv[i]);
    if (i < DIMC * HID) wout_bf[i] = f2bf(wout[i]);
}

// ---------------------------------------------------------------- kernel 1
// qkv[b] (384x8192, bf16) = Wqkv (384x256, bf16) @ x[b] (256x8192, f32)
// Tiles: 128(M) x 128(N), K-steps of 32.  8 waves; wave w owns N-substrip w.
__global__ __launch_bounds__(256) void qkv_gemm(
    const float* __restrict__ x, const unsigned short* __restrict__ wq,
    unsigned short* __restrict__ qkv) {
    __shared__ unsigned short As[128 * 32];   // [m][k]
    __shared__ unsigned short Bs[128 * 32];   // transposed: [n][k]
    const int b = blockIdx.z, m0 = blockIdx.y * 128, n0 = blockIdx.x * 128;
    const int t = threadIdx.x, lane = t & 31, wv = t >> 5;
    const int kbA = (lane >> 4) * 16, kbB = (lane >> 4) * 16, ln = lane & 15;
    const float* xb = x + (size_t)b * DIMC * NPOS;
    v8f acc[8] = {};
    for (int kk = 0; kk < DIMC; kk += 32) {
        { // stage A: 128x32 bf16, pure copy -> async global->LDS (2x16B/lane)
            int row = t >> 1, ch = (t & 1) * 16;
            const unsigned short* src = wq + (size_t)(m0 + row) * DIMC + kk + ch;
            unsigned short* dst = As + row * 32 + ch;
            async_copy16(src, dst);
            async_copy16(src + 8, dst + 8);
        }
        { // stage B: x rows kk..kk+31, cols n0..n0+127 (f32 -> bf16, transpose into LDS)
            int r = t >> 3, c0 = (t & 7) * 16;
            const float* src = xb + (size_t)(kk + r) * NPOS + n0 + c0;
            if (kk + 32 < DIMC) __builtin_prefetch(src + (size_t)32 * NPOS, 0, 1);
#pragma unroll
            for (int j = 0; j < 16; j++) Bs[(c0 + j) * 32 + r] = f2bf(src[j]);
        }
        wait_async0();
        __syncthreads();
        BF16x16 bop;
        {
            const uint4* p = (const uint4*)(Bs + (wv * 16 + ln) * 32 + kbB);
            bop.q[0] = p[0]; bop.q[1] = p[1];
        }
#pragma unroll
        for (int ms = 0; ms < 8; ms++) {
            BF16x16 aop;
            const unsigned short* ap = As + (ms * 16 + ln) * 32;
            aop.q[0] = *(const uint4*)(ap + kbA);
            aop.q[1] = *(const uint4*)(ap + kbA + 8);
            acc[ms] = __builtin_amdgcn_wmma_f32_16x16x32_bf16(
                false, aop.v, false, bop.v, (short)0, acc[ms], false, false);
        }
        __syncthreads();
    }
    const int rowoff = (lane >> 4) ? 8 : 0;
#pragma unroll
    for (int ms = 0; ms < 8; ms++)
#pragma unroll
        for (int r = 0; r < 8; r++) {
            int row = m0 + ms * 16 + r + rowoff;
            qkv[((size_t)b * O3 + row) * NPOS + n0 + wv * 16 + ln] = f2bf(acc[ms][r]);
        }
}

// ---------------------------------------------------------------- kernel 2
// Per k-row (b,h,d): max over N and sum of exp(k - max).
__global__ __launch_bounds__(256) void softmax_stats(
    const unsigned short* __restrict__ qkv, float* __restrict__ stats) {
    __shared__ float red[256];
    const int row = blockIdx.x;            // 0 .. B*HID-1
    const int b = row >> 7, hd = row & 127;
    const unsigned short* kr = qkv + ((size_t)b * O3 + HID + hd) * NPOS;
    const int t = threadIdx.x;
    float m = -3.4e38f;
    for (int i = t; i < NPOS; i += 256) m = fmaxf(m, bf2f(kr[i]));
    red[t] = m; __syncthreads();
    for (int s = 128; s > 0; s >>= 1) { if (t < s) red[t] = fmaxf(red[t], red[t + s]); __syncthreads(); }
    m = red[0]; __syncthreads();
    float sum = 0.f;
    for (int i = t; i < NPOS; i += 256) sum += __expf(bf2f(kr[i]) - m);
    red[t] = sum; __syncthreads();
    for (int s = 128; s > 0; s >>= 1) { if (t < s) red[t] += red[t + s]; __syncthreads(); }
    if (t == 0) { stats[row * 2] = m; stats[row * 2 + 1] = red[0]; }
}

// ---------------------------------------------------------------- kernel 3
// ctx[b,h] (32x32) = softmax(k) @ v^T ; exp applied while packing A operands.
__global__ __launch_bounds__(256) void context_kernel(
    const unsigned short* __restrict__ qkv, const float* __restrict__ stats,
    unsigned short* __restrict__ ctx) {
    __shared__ float red[8 * 32 * 32];     // per-wave partial ctx
    const int bh = blockIdx.x, b = bh >> 2, h = bh & 3;
    const int t = threadIdx.x, lane = t & 31, wv = t >> 5;
    const int kbA = (lane >> 4) * 16, kbB = (lane >> 4) * 16, ln = lane & 15;
    const unsigned short* kb = qkv + ((size_t)b * O3 + HID + h * DHEAD) * NPOS;
    const unsigned short* vb = qkv + ((size_t)b * O3 + 2 * HID + h * DHEAD) * NPOS;
    float mx[2];
    mx[0] = stats[(b * HID + h * DHEAD + ln) * 2];
    mx[1] = stats[(b * HID + h * DHEAD + 16 + ln) * 2];
    v8f acc[2][2] = {};
    for (int it = 0; it < 32; ++it) {      // each wave covers N/8 = 1024
        int nb = wv * 1024 + it * 32;
        BF16x16 bop[2];
#pragma unroll
        for (int tj = 0; tj < 2; tj++) {   // B = v^T: column e, K=n contiguous
            const uint4* p = (const uint4*)(vb + (size_t)(tj * 16 + ln) * NPOS + nb + kbB);
            bop[tj].q[0] = p[0]; bop[tj].q[1] = p[1];
        }
#pragma unroll
        for (int ti = 0; ti < 2; ti++) {
            BF16x16 raw, aop;
            const unsigned short* kp = kb + (size_t)(ti * 16 + ln) * NPOS + nb;
            raw.q[0] = *(const uint4*)(kp + kbA);
            raw.q[1] = *(const uint4*)(kp + kbA + 8);
#pragma unroll
            for (int j = 0; j < 16; j++) aop.u[j] = f2bf(__expf(bf2f(raw.u[j]) - mx[ti]));
#pragma unroll
            for (int tj = 0; tj < 2; tj++)
                acc[ti][tj] = __builtin_amdgcn_wmma_f32_16x16x32_bf16(
                    false, aop.v, false, bop[tj].v, (short)0, acc[ti][tj], false, false);
        }
    }
    const int rowoff = (lane >> 4) ? 8 : 0;
#pragma unroll
    for (int ti = 0; ti < 2; ti++)
#pragma unroll
        for (int tj = 0; tj < 2; tj++)
#pragma unroll
            for (int r = 0; r < 8; r++) {
                int d = ti * 16 + r + rowoff, e = tj * 16 + ln;
                red[wv * 1024 + d * 32 + e] = acc[ti][tj][r];
            }
    __syncthreads();
    for (int i = t; i < 1024; i += 256) {  // reduce 8 waves, normalize by sumexp
        float s = 0.f;
#pragma unroll
        for (int w = 0; w < 8; w++) s += red[w * 1024 + i];
        int d = i >> 5;
        float inv = 1.f / stats[(b * HID + h * DHEAD + d) * 2 + 1];
        ctx[(size_t)bh * 1024 + i] = f2bf(s * inv);
    }
}

// ---------------------------------------------------------------- kernel 4
// M_b[:, h*32+d] = sum_e wout[:, h*32+e] * ctx[b,h][d,e]  (K=32 -> 1 WMMA/tile)
__global__ __launch_bounds__(256) void mmat_kernel(
    const unsigned short* __restrict__ wout, const unsigned short* __restrict__ ctx,
    unsigned short* __restrict__ Mm) {
    const int bh = blockIdx.x, b = bh >> 2, h = bh & 3;
    const int t = threadIdx.x, lane = t & 31, wv = t >> 5;
    const int kbA = (lane >> 4) * 16, kbB = (lane >> 4) * 16, ln = lane & 15;
    const unsigned short* cx = ctx + (size_t)bh * 1024;
    const int rowoff = (lane >> 4) ? 8 : 0;
#pragma unroll
    for (int mi = 0; mi < 2; ++mi) {
        int mtile = wv * 2 + mi;           // 16 M-tiles over 8 waves
#pragma unroll
        for (int dt = 0; dt < 2; ++dt) {
            BF16x16 aop, bop;
            const unsigned short* ap = wout + (size_t)(mtile * 16 + ln) * HID + h * DHEAD;
            aop.q[0] = *(const uint4*)(ap + kbA);
            aop.q[1] = *(const uint4*)(ap + kbA + 8);
            const uint4* bp = (const uint4*)(cx + (dt * 16 + ln) * DHEAD + kbB);
            bop.q[0] = bp[0]; bop.q[1] = bp[1];
            v8f acc = {};
            acc = __builtin_amdgcn_wmma_f32_16x16x32_bf16(
                false, aop.v, false, bop.v, (short)0, acc, false, false);
#pragma unroll
            for (int r = 0; r < 8; r++) {
                int c = mtile * 16 + r + rowoff, d = dt * 16 + ln;
                Mm[((size_t)b * DIMC + c) * HID + h * DHEAD + d] = f2bf(acc[r]);
            }
        }
    }
}

// ---------------------------------------------------------------- kernel 5
// out[b] (256x8192, f32) = M_b (256x128, bf16) @ q[b] (128x8192, bf16) + b_out
__global__ __launch_bounds__(256) void out_gemm(
    const unsigned short* __restrict__ Mm, const unsigned short* __restrict__ qkv,
    const float* __restrict__ bout, float* __restrict__ out) {
    __shared__ unsigned short As[128 * 32];
    __shared__ unsigned short Bs[128 * 32];   // [n][k]
    const int b = blockIdx.z, m0 = blockIdx.y * 128, n0 = blockIdx.x * 128;
    const int t = threadIdx.x, lane = t & 31, wv = t >> 5;
    const int kbA = (lane >> 4) * 16, kbB = (lane >> 4) * 16, ln = lane & 15;
    const unsigned short* Ab = Mm + (size_t)b * DIMC * HID;
    const unsigned short* Qb = qkv + (size_t)b * O3 * NPOS;   // q = rows 0..127
    v8f acc[8] = {};
    for (int kk = 0; kk < HID; kk += 32) {
        { // stage A: pure copy -> async global->LDS
            int row = t >> 1, ch = (t & 1) * 16;
            const unsigned short* src = Ab + (size_t)(m0 + row) * HID + kk + ch;
            unsigned short* dst = As + row * 32 + ch;
            async_copy16(src, dst);
            async_copy16(src + 8, dst + 8);
        }
        { // stage B: bf16 q rows, transpose into LDS
            int r = t >> 3, c0 = (t & 7) * 16;
            const unsigned short* src = Qb + (size_t)(kk + r) * NPOS + n0 + c0;
            if (kk + 32 < HID) __builtin_prefetch(src + (size_t)32 * NPOS, 0, 1);
#pragma unroll
            for (int j = 0; j < 16; j++) Bs[(c0 + j) * 32 + r] = src[j];
        }
        wait_async0();
        __syncthreads();
        BF16x16 bop;
        {
            const uint4* p = (const uint4*)(Bs + (wv * 16 + ln) * 32 + kbB);
            bop.q[0] = p[0]; bop.q[1] = p[1];
        }
#pragma unroll
        for (int ms = 0; ms < 8; ms++) {
            BF16x16 aop;
            const unsigned short* ap = As + (ms * 16 + ln) * 32;
            aop.q[0] = *(const uint4*)(ap + kbA);
            aop.q[1] = *(const uint4*)(ap + kbA + 8);
            acc[ms] = __builtin_amdgcn_wmma_f32_16x16x32_bf16(
                false, aop.v, false, bop.v, (short)0, acc[ms], false, false);
        }
        __syncthreads();
    }
    const int rowoff = (lane >> 4) ? 8 : 0;
#pragma unroll
    for (int ms = 0; ms < 8; ms++)
#pragma unroll
        for (int r = 0; r < 8; r++) {
            int row = m0 + ms * 16 + r + rowoff;
            out[((size_t)b * DIMC + row) * NPOS + n0 + wv * 16 + ln] = acc[ms][r] + bout[row];
        }
}

// ---------------------------------------------------------------- launcher
extern "C" void kernel_launch(void* const* d_in, const int* in_sizes, int n_in,
                              void* d_out, int out_size, void* d_ws, size_t ws_size,
                              hipStream_t stream) {
    const float* x    = (const float*)d_in[0];
    const float* wqkv = (const float*)d_in[1];
    const float* wout = (const float*)d_in[2];
    const float* bout = (const float*)d_in[3];
    float* out = (float*)d_out;

    char* ws = (char*)d_ws;
    size_t off = 0;
    unsigned short* qkv_bf = (unsigned short*)(ws + off); off += (size_t)B_ * O3 * NPOS * 2;
    float*          stats  = (float*)(ws + off);          off += (size_t)B_ * HID * 2 * 4;
    unsigned short* ctx    = (unsigned short*)(ws + off); off += (size_t)B_ * HEADS * 32 * 32 * 2;
    unsigned short* Mm     = (unsigned short*)(ws + off); off += (size_t)B_ * DIMC * HID * 2;
    unsigned short* wq_bf  = (unsigned short*)(ws + off); off += (size_t)O3 * DIMC * 2;
    unsigned short* wo_bf  = (unsigned short*)(ws + off); off += (size_t)DIMC * HID * 2;

    cvt_weights   <<<(O3 * DIMC + 255) / 256, 256, 0, stream>>>(wqkv, wout, wq_bf, wo_bf);
    qkv_gemm      <<<dim3(NPOS / 128, O3 / 128, B_), 256, 0, stream>>>(x, wq_bf, qkv_bf);
    softmax_stats <<<B_ * HID, 256, 0, stream>>>(qkv_bf, stats);
    context_kernel<<<B_ * HEADS, 256, 0, stream>>>(qkv_bf, stats, ctx);
    mmat_kernel   <<<B_ * HEADS, 256, 0, stream>>>(wo_bf, ctx, Mm);
    out_gemm      <<<dim3(NPOS / 128, DIMC / 128, B_), 256, 0, stream>>>(Mm, qkv_bf, bout, out);
}
